// MultiHeadAttention_65962107732817
// MI455X (gfx1250) — matrix operand order repaired and checked by reading, not compiled, not run on hardware
//
#include <hip/hip_runtime.h>

typedef _Float16 half_t;
typedef __attribute__((ext_vector_type(16))) _Float16 v16h;
typedef __attribute__((ext_vector_type(8)))  _Float16 v8h;
typedef __attribute__((ext_vector_type(8)))  float    v8f;

constexpr int Bc = 4, Tc = 2048, Dc = 1024, Hc = 16, HDc = 64;
// fold 1/sqrt(HD) * log2(e) into Q so attention uses exp2 directly
constexpr float QSCALE = 0.125f * 1.44269504088896f;

// ---------- fragment helpers ----------
// A 16x32 f16 fragment, contiguous-k order used here (AMD's table differs): lanes 0-15 row M=lane, K=0..15;
// lanes 16-31 row M=lane-16, K=16..31. Caller folds the
// per-half-wave +16 K offset into p; elements 0..7 at p, 8..15 at p+8.
template <typename P>
__device__ __forceinline__ v16h load_frag_a(const P* p) {
  const v8h lo = *reinterpret_cast<const v8h*>(p);
  const v8h hi = *reinterpret_cast<const v8h*>(p + 8);
  v16h r;
#pragma unroll
  for (int i = 0; i < 8; ++i) { r[i] = lo[i]; r[i + 8] = hi[i]; }
  return r;
}

// B 32x16 fragment, same k order as A (not AMD's table): lane holds column N (=lane&15) with 16 contiguous
// K values (lanes 0-15: K=0..15, lanes 16-31: K=16..31; +16 offset folded in).
template <typename P>
__device__ __forceinline__ v16h load_frag_b(const P* p) {
  const v8h lo = *reinterpret_cast<const v8h*>(p);
  const v8h hi = *reinterpret_cast<const v8h*>(p + 8);
  v16h r;
#pragma unroll
  for (int i = 0; i < 8; ++i) { r[i] = lo[i]; r[i + 8] = hi[i]; }
  return r;
}

__device__ __forceinline__ v8f wmma_f16(v16h a, v16h b, v8f c) {
  return __builtin_amdgcn_wmma_f32_16x16x32_f16(false, a, false, b, (short)0, c,
                                                false, false);
}

// ---------- conversion kernels ----------
__global__ void __launch_bounds__(256)
cvt_f32_to_f16(const float* __restrict__ src, half_t* __restrict__ dst) {
  const int i = blockIdx.x * blockDim.x + threadIdx.x;
  const float4 v = reinterpret_cast<const float4*>(src)[i];
  half_t* d = dst + 4 * (size_t)i;
  d[0] = (half_t)v.x; d[1] = (half_t)v.y; d[2] = (half_t)v.z; d[3] = (half_t)v.w;
}

// src: [H][D][HD] f32  ->  dst: [H][HD][D] f16 (transposed per head)
__global__ void __launch_bounds__(256)
cvt_w_transpose(const float* __restrict__ src, half_t* __restrict__ dst) {
  const int i = blockIdx.x * blockDim.x + threadIdx.x;
  const int h = i / (Dc * HDc);
  const int rem = i % (Dc * HDc);
  const int d = rem / HDc, hd = rem % HDc;
  dst[((size_t)h * HDc + hd) * Dc + d] = (half_t)src[i];
}

// ---------- QKV projection: Q/K/V[b,h] = x[b] @ W[h], LDS-tiled ----------
// grid: (T/64, B*H), block = 256 (8 waves). Block covers 64 rows x all 12
// N-tiles. Wave (mtile = wid&3, grp = wid>>2) owns 6 N-jobs: grp0 -> Q nt0-3 +
// K nt0-1, grp1 -> K nt2-3 + V nt0-3. B-row for job j is simply 16*j + ln.
constexpr int XP = 40;  // padded LDS row stride (halfs): 20 banks, conflict-free
constexpr int WP = 40;

__global__ void __launch_bounds__(256)
qkv_proj_kernel(const half_t* __restrict__ xh, const half_t* __restrict__ Wqt,
                const half_t* __restrict__ Wkt, const half_t* __restrict__ Wvt,
                half_t* __restrict__ Qb, half_t* __restrict__ Kb,
                half_t* __restrict__ Vt) {
  constexpr int XBUF = 64 * XP;           // 2560 halfs
  constexpr int WBUF = 192 * WP;          // 7680 halfs
  constexpr int CBUF = XBUF + WBUF;
  __shared__ __align__(16) half_t smem[2 * CBUF];

  const int tid  = threadIdx.x;
  const int lane = tid & 31;
  const int wid  = tid >> 5;
  const int ln = lane & 15;
  const int g  = lane >> 4;
  const int mtile = wid & 3;
  const int grp   = wid >> 2;
  const int j0    = grp * 6;
  const int tb = blockIdx.x * 64;
  const int bh = blockIdx.y;
  const int b = bh >> 4, h = bh & 15;

  // staging: every thread moves 1 v8h of x and 1 v8h from each weight matrix
  const int xr = tid >> 2, xc = (tid & 3) * 8;   // x rows 0..63, cols 0..31
  const half_t* xsrc = xh + ((size_t)b * Tc + tb + xr) * Dc + xc;
  const size_t woff = ((size_t)h * HDc + xr) * Dc + xc;
  const half_t* wq = Wqt + woff;
  const half_t* wk = Wkt + woff;
  const half_t* wv = Wvt + woff;

  v8f acc[6];
  const v8f vzero = {};
#pragma unroll
  for (int i = 0; i < 6; ++i) acc[i] = vzero;

  auto compute = [&](const half_t* Xl, const half_t* Wl) {
    const v16h af = load_frag_a(&Xl[(mtile * 16 + ln) * XP + 16 * g]);
    v16h bf = load_frag_b(&Wl[(16 * j0 + ln) * WP + 16 * g]);
#pragma unroll
    for (int i = 0; i < 6; ++i) {
      v16h bfn;
      if (i < 5)
        bfn = load_frag_b(&Wl[(16 * (j0 + i + 1) + ln) * WP + 16 * g]);
      acc[i] = wmma_f16(af, bf, acc[i]);
      if (i < 5) bf = bfn;
    }
  };

  // prologue: stage k0 = 0 into buffer 0
  *reinterpret_cast<v8h*>(&smem[xr * XP + xc]) =
      *reinterpret_cast<const v8h*>(xsrc);
  *reinterpret_cast<v8h*>(&smem[XBUF + xr * WP + xc]) =
      *reinterpret_cast<const v8h*>(wq);
  *reinterpret_cast<v8h*>(&smem[XBUF + (64 + xr) * WP + xc]) =
      *reinterpret_cast<const v8h*>(wk);
  *reinterpret_cast<v8h*>(&smem[XBUF + (128 + xr) * WP + xc]) =
      *reinterpret_cast<const v8h*>(wv);
  __syncthreads();

  for (int it = 0; it < 31; ++it) {
    const int p = it & 1;
    const int k1 = (it + 1) * 32;

    // issue next chunk's global loads early (latency hidden under compute)
    const v8h nx = *reinterpret_cast<const v8h*>(xsrc + k1);
    const v8h nq = *reinterpret_cast<const v8h*>(wq + k1);
    const v8h nk = *reinterpret_cast<const v8h*>(wk + k1);
    const v8h nv = *reinterpret_cast<const v8h*>(wv + k1);

    compute(smem + p * CBUF, smem + p * CBUF + XBUF);

    half_t* Xn = smem + (p ^ 1) * CBUF;
    half_t* Wn = Xn + XBUF;
    *reinterpret_cast<v8h*>(&Xn[xr * XP + xc]) = nx;
    *reinterpret_cast<v8h*>(&Wn[xr * WP + xc]) = nq;
    *reinterpret_cast<v8h*>(&Wn[(64 + xr) * WP + xc]) = nk;
    *reinterpret_cast<v8h*>(&Wn[(128 + xr) * WP + xc]) = nv;
    __syncthreads();
  }
  compute(smem + CBUF, smem + CBUF + XBUF);  // it = 31, buffer 1

  // epilogue: Q (pre-scaled),K row-major [bh][t][hd]; V transposed [bh][hd][t]
  const int t0 = tb + mtile * 16;
  if (grp == 0) {
    half_t* dq = Qb + ((size_t)bh * Tc + t0) * HDc;
#pragma unroll
    for (int nt = 0; nt < 4; ++nt)
#pragma unroll
      for (int r = 0; r < 8; ++r)
        dq[(size_t)(r + 8 * g) * HDc + nt * 16 + ln] =
            (half_t)(acc[nt][r] * QSCALE);
    half_t* dk = Kb + ((size_t)bh * Tc + t0) * HDc;
#pragma unroll
    for (int i = 0; i < 2; ++i)
#pragma unroll
      for (int r = 0; r < 8; ++r)
        dk[(size_t)(r + 8 * g) * HDc + i * 16 + ln] = (half_t)acc[4 + i][r];
  } else {
    half_t* dk = Kb + ((size_t)bh * Tc + t0) * HDc;
#pragma unroll
    for (int i = 0; i < 2; ++i)
#pragma unroll
      for (int r = 0; r < 8; ++r)
        dk[(size_t)(r + 8 * g) * HDc + (2 + i) * 16 + ln] = (half_t)acc[i][r];
#pragma unroll
    for (int nt = 0; nt < 4; ++nt) {
      half_t* dst = Vt + ((size_t)bh * HDc + nt * 16 + ln) * Tc + t0 + 8 * g;
      v8h pk;
#pragma unroll
      for (int r = 0; r < 8; ++r) pk[r] = (half_t)acc[2 + nt][r];
      *reinterpret_cast<v8h*>(dst) = pk;
    }
  }
}

// ---------- flash attention, K/V chunks shared across block via LDS ----------
// grid: (T/128, B*H), block = 256 (8 waves = 8 query tiles of the same (b,h)).
// Double-buffered K/V staging, one barrier per chunk, exp2-domain softmax.
constexpr int KP = 72;  // 32x64 K-chunk rows, padded (36 banks stride)
constexpr int VP = 40;  // 64x32 V-chunk rows, padded
constexpr int PP = 40;  // 16x32 per-wave P scratch, padded

__global__ void __launch_bounds__(256)
attn_kernel(const half_t* __restrict__ Qb, const half_t* __restrict__ Kb,
            const half_t* __restrict__ Vt, half_t* __restrict__ concat) {
  constexpr int KBUF = 32 * KP;            // 2304 halfs
  constexpr int VBUF = 64 * VP;            // 2560 halfs
  constexpr int CBUF = KBUF + VBUF;
  __shared__ __align__(16) half_t smem[2 * CBUF + 8 * 16 * PP];

  const int tid  = threadIdx.x;
  const int lane = tid & 31;
  const int wid  = tid >> 5;
  const int ln = lane & 15;
  const int g  = lane >> 4;
  half_t* Pl = smem + 2 * CBUF + wid * (16 * PP);
  const int bh = blockIdx.y;
  const int b = bh >> 4, h = bh & 15;
  const int t0 = (blockIdx.x * 8 + wid) * 16;

  // preload Q A-fragments (K-dim = hd, two 32-wide steps); Q is pre-scaled
  const half_t* Qp = Qb + ((size_t)bh * Tc + t0 + ln) * HDc + 16 * g;
  const v16h qf0 = load_frag_a(Qp);
  const v16h qf1 = load_frag_a(Qp + 32);

  const half_t* Kbase = Kb + (size_t)bh * Tc * HDc;
  const half_t* Vbase = Vt + (size_t)bh * HDc * Tc;

  // staging coordinates: 2048 halfs each, 8 per thread
  const int kr = tid >> 3, kc = (tid & 7) * 8;   // K rows 0..31, cols 0..63
  const int vr = tid >> 2, vc = (tid & 3) * 8;   // V rows 0..63, cols 0..31

  float m[8], l[8];
  v8f acc[4];
  const v8f vzero = {};
#pragma unroll
  for (int r = 0; r < 8; ++r) { m[r] = -1e30f; l[r] = 0.f; }
#pragma unroll
  for (int n = 0; n < 4; ++n) acc[n] = vzero;

  auto compute = [&](const half_t* Klds, const half_t* Vlds) {
    // scores: K-fragments hoisted two at a time (bounded liveness)
    v8f sc0 = vzero, sc1 = vzero;
    {
      const v16h ka = load_frag_b(&Klds[ln * KP + 16 * g]);
      const v16h kb = load_frag_b(&Klds[ln * KP + 32 + 16 * g]);
      sc0 = wmma_f16(qf0, ka, sc0);
      sc0 = wmma_f16(qf1, kb, sc0);
    }
    {
      const v16h ka = load_frag_b(&Klds[(16 + ln) * KP + 16 * g]);
      const v16h kb = load_frag_b(&Klds[(16 + ln) * KP + 32 + 16 * g]);
      sc1 = wmma_f16(qf0, ka, sc1);
      sc1 = wmma_f16(qf1, kb, sc1);
    }

    // online softmax in exp2 domain; sc0/sc1 overwritten with probabilities
    float mc[8], corr[8], rs[8];
#pragma unroll
    for (int r = 0; r < 8; ++r) mc[r] = fmaxf(sc0[r], sc1[r]);
#pragma unroll
    for (int mk = 1; mk < 16; mk <<= 1)
#pragma unroll
      for (int r = 0; r < 8; ++r)
        mc[r] = fmaxf(mc[r], __shfl_xor(mc[r], mk, 32));
#pragma unroll
    for (int r = 0; r < 8; ++r) {
      const float mn = fmaxf(m[r], mc[r]);
      corr[r] = exp2f(m[r] - mn);
      m[r] = mn;
      sc0[r] = exp2f(sc0[r] - mn);
      sc1[r] = exp2f(sc1[r] - mn);
      rs[r] = sc0[r] + sc1[r];
    }
#pragma unroll
    for (int mk = 1; mk < 16; mk <<= 1)
#pragma unroll
      for (int r = 0; r < 8; ++r)
        rs[r] += __shfl_xor(rs[r], mk, 32);
#pragma unroll
    for (int r = 0; r < 8; ++r) l[r] = l[r] * corr[r] + rs[r];
#pragma unroll
    for (int n = 0; n < 4; ++n)
#pragma unroll
      for (int r = 0; r < 8; ++r) acc[n][r] *= corr[r];

    // D-layout -> per-wave LDS scratch -> A-fragment (same-wave DS in order)
#pragma unroll
    for (int r = 0; r < 8; ++r) {
      const int row = r + 8 * g;
      Pl[row * PP + ln]      = (half_t)sc0[r];
      Pl[row * PP + 16 + ln] = (half_t)sc1[r];
    }
    const v16h pa = load_frag_a(Pl + ln * PP + 16 * g);

    // acc += P @ V with single-step pipelined V-fragments
    v16h vb = load_frag_b(&Vlds[ln * VP + 16 * g]);
#pragma unroll
    for (int nt = 0; nt < 4; ++nt) {
      v16h vbn;
      if (nt < 3)
        vbn = load_frag_b(&Vlds[((nt + 1) * 16 + ln) * VP + 16 * g]);
      acc[nt] = wmma_f16(pa, vb, acc[nt]);
      if (nt < 3) vb = vbn;
    }
  };

  // prologue: stage chunk 0 into buffer 0
  *reinterpret_cast<v8h*>(&smem[kr * KP + kc]) =
      *reinterpret_cast<const v8h*>(Kbase + (size_t)kr * HDc + kc);
  *reinterpret_cast<v8h*>(&smem[KBUF + vr * VP + vc]) =
      *reinterpret_cast<const v8h*>(Vbase + (size_t)vr * Tc + vc);
  __syncthreads();

  for (int it = 0; it < Tc / 32 - 1; ++it) {
    const int p = it & 1;
    const int s1 = (it + 1) * 32;

    const v8h nk =
        *reinterpret_cast<const v8h*>(Kbase + (size_t)(s1 + kr) * HDc + kc);
    const v8h nv =
        *reinterpret_cast<const v8h*>(Vbase + (size_t)vr * Tc + s1 + vc);

    compute(smem + p * CBUF, smem + p * CBUF + KBUF);

    half_t* Kn = smem + (p ^ 1) * CBUF;
    *reinterpret_cast<v8h*>(&Kn[kr * KP + kc]) = nk;
    *reinterpret_cast<v8h*>(&Kn[KBUF + vr * VP + vc]) = nv;
    __syncthreads();
  }
  compute(smem + CBUF, smem + CBUF + KBUF);  // final chunk, buffer 1

  // finalize + store concat[b][t][h*64+hd] in f16
  float inv[8];
#pragma unroll
  for (int r = 0; r < 8; ++r) inv[r] = 1.0f / l[r];
  half_t* cp = concat + ((size_t)b * Tc + t0) * Dc + h * HDc;
#pragma unroll
  for (int nt = 0; nt < 4; ++nt)
#pragma unroll
    for (int r = 0; r < 8; ++r)
      cp[(size_t)(r + 8 * g) * Dc + nt * 16 + ln] = (half_t)(acc[nt][r] * inv[r]);
}

// ---------- output projection: out = concat @ Wo^T, LDS-tiled ----------
// grid: (B*T/128, D/64), block = 256. Wave owns one M-tile, 4 N-tiles.
constexpr int BP = 40;

__global__ void __launch_bounds__(256)
out_proj_kernel(const half_t* __restrict__ concat,
                const half_t* __restrict__ Woh, float* __restrict__ out) {
  constexpr int ABUF = 128 * XP;          // 5120 halfs
  constexpr int BBUF = 64 * BP;           // 2560 halfs
  constexpr int CBUF = ABUF + BBUF;
  __shared__ __align__(16) half_t smem[2 * CBUF];

  const int tid  = threadIdx.x;
  const int lane = tid & 31;
  const int wid  = tid >> 5;
  const int ln = lane & 15;
  const int g  = lane >> 4;
  const int m0 = blockIdx.x * 128;
  const int n0 = blockIdx.y * 64;

  const int ar = tid >> 1, ac = (tid & 1) * 16;
  const half_t* asrc = concat + (size_t)(m0 + ar) * Dc + ac;
  const int br = tid >> 2, bcc = (tid & 3) * 8;
  const half_t* bsrc = Woh + (size_t)(n0 + br) * Dc + bcc;  // B[k=e][n=d]=Wo[n][k]

  v8f acc[4];
  const v8f vzero = {};
#pragma unroll
  for (int n = 0; n < 4; ++n) acc[n] = vzero;

  auto compute = [&](const half_t* Al, const half_t* Bl) {
    const v16h af = load_frag_a(&Al[(wid * 16 + ln) * XP + 16 * g]);
    v16h bf = load_frag_b(&Bl[ln * BP + 16 * g]);
#pragma unroll
    for (int nt = 0; nt < 4; ++nt) {
      v16h bfn;
      if (nt < 3)
        bfn = load_frag_b(&Bl[((nt + 1) * 16 + ln) * BP + 16 * g]);
      acc[nt] = wmma_f16(af, bf, acc[nt]);
      if (nt < 3) bf = bfn;
    }
  };

  // prologue: stage k0 = 0 into buffer 0
  *reinterpret_cast<v8h*>(&smem[ar * XP + ac]) =
      *reinterpret_cast<const v8h*>(asrc);
  *reinterpret_cast<v8h*>(&smem[ar * XP + ac + 8]) =
      *reinterpret_cast<const v8h*>(asrc + 8);
  *reinterpret_cast<v8h*>(&smem[ABUF + br * BP + bcc]) =
      *reinterpret_cast<const v8h*>(bsrc);
  __syncthreads();

  for (int it = 0; it < 31; ++it) {
    const int p = it & 1;
    const int k1 = (it + 1) * 32;

    const v8h na0 = *reinterpret_cast<const v8h*>(asrc + k1);
    const v8h na1 = *reinterpret_cast<const v8h*>(asrc + k1 + 8);
    const v8h nb0 = *reinterpret_cast<const v8h*>(bsrc + k1);

    compute(smem + p * CBUF, smem + p * CBUF + ABUF);

    half_t* An = smem + (p ^ 1) * CBUF;
    *reinterpret_cast<v8h*>(&An[ar * XP + ac]) = na0;
    *reinterpret_cast<v8h*>(&An[ar * XP + ac + 8]) = na1;
    *reinterpret_cast<v8h*>(&An[ABUF + br * BP + bcc]) = nb0;
    __syncthreads();
  }
  compute(smem + CBUF, smem + CBUF + ABUF);  // it = 31, buffer 1

  float* dst = out + (size_t)(m0 + wid * 16) * Dc + n0;
#pragma unroll
  for (int nt = 0; nt < 4; ++nt)
#pragma unroll
    for (int r = 0; r < 8; ++r)
      dst[(size_t)(r + 8 * g) * Dc + nt * 16 + ln] = acc[nt][r];
}

extern "C" void kernel_launch(void* const* d_in, const int* in_sizes, int n_in,
                              void* d_out, int out_size, void* d_ws,
                              size_t ws_size, hipStream_t stream) {
  const float* x  = (const float*)d_in[0];
  const float* Wq = (const float*)d_in[1];
  const float* Wk = (const float*)d_in[2];
  const float* Wv = (const float*)d_in[3];
  const float* Wo = (const float*)d_in[4];
  float* out = (float*)d_out;

  char* ws = (char*)d_ws;
  size_t off = 0;
  auto carve = [&](size_t elems) {
    half_t* p = (half_t*)(ws + off);
    off += (elems * sizeof(half_t) + 255) & ~(size_t)255;
    return p;
  };
  const size_t nX  = (size_t)Bc * Tc * Dc;        // 8M
  const size_t nW  = (size_t)Hc * Dc * HDc;       // 1M per matrix
  const size_t nWo = (size_t)Dc * Dc;             // 1M
  const size_t nQ  = (size_t)Bc * Hc * Tc * HDc;  // 8M

  half_t* xh   = carve(nX);
  half_t* Wqt  = carve(nW);
  half_t* Wkt  = carve(nW);
  half_t* Wvt  = carve(nW);
  half_t* Woh  = carve(nWo);
  half_t* Qb   = carve(nQ);
  half_t* Kb   = carve(nQ);
  half_t* Vt   = carve(nQ);
  half_t* conc = carve(nX);

  cvt_f32_to_f16<<<(int)(nX / (4 * 256)), 256, 0, stream>>>(x, xh);
  cvt_w_transpose<<<(int)(nW / 256), 256, 0, stream>>>(Wq, Wqt);
  cvt_w_transpose<<<(int)(nW / 256), 256, 0, stream>>>(Wk, Wkt);
  cvt_w_transpose<<<(int)(nW / 256), 256, 0, stream>>>(Wv, Wvt);
  cvt_f32_to_f16<<<(int)(nWo / (4 * 256)), 256, 0, stream>>>(Wo, Woh);

  qkv_proj_kernel<<<dim3(Tc / 64, Bc * Hc), 256, 0, stream>>>(xh, Wqt, Wkt,
                                                              Wvt, Qb, Kb, Vt);
  attn_kernel<<<dim3(Tc / 128, Bc * Hc), 256, 0, stream>>>(Qb, Kb, Vt, conc);
  out_proj_kernel<<<dim3((Bc * Tc) / 128, Dc / 64), 256, 0, stream>>>(conc, Woh,
                                                                      out);
}
